// PairWeightedAveraging_22179211116939
// MI455X (gfx1250) — matrix operand order repaired and checked by reading, not compiled, not run on hardware
//
#include <hip/hip_runtime.h>
#include <hip/hip_bf16.h>
#include <math.h>

// ---------------------------------------------------------------------------
// Problem constants (B=1): S=256, N=768, CM=64, CZ=128, H=8, CH=32
// ---------------------------------------------------------------------------
#define S_DIM   256
#define N_RES   768
#define CM_DIM  64
#define CZ_DIM  128
#define H_DIM   8
#define CH_DIM  32
#define ROWS_M  (S_DIM * N_RES)          // 196608
#define ROWS_Z  (N_RES * N_RES)          // 589824
#define HC_DIM  (H_DIM * CH_DIM)         // 256
#define VN_DIM  (S_DIM * CH_DIM)         // 8192 columns of attention GEMM

typedef __attribute__((ext_vector_type(16))) _Float16 v16h;
typedef __attribute__((ext_vector_type(8)))  float    v8f;

union HF2 { uint4 u[2]; v16h h; };

// A fragment (16x32 f16): lane holds row M=(lane&15); K chunks kg*8..kg*8+7 and
// 16+kg*8..16+kg*8+7 where kg = lane>>4.  Two 16B LDS loads.
__device__ __forceinline__ v16h load_frag_a(const _Float16* base) {
    HF2 t;
    t.u[0] = *(const uint4*)(base);
    t.u[1] = *(const uint4*)(base + 16);
    return t.h;
}
// B fragment (32x16 f16): lane holds column N=(lane&15); K = kg*8..kg*8+7 and
// 16+kg*8..16+kg*8+7 where kg = lane>>4 (strided LDS gather).
__device__ __forceinline__ v16h load_frag_b(const _Float16* base, int ld, int kg) {
    v16h t;
    #pragma unroll
    for (int s = 0; s < 8; ++s) {
        t[s]     = base[(kg * 8 + s) * ld];
        t[s + 8] = base[(16 + kg * 8 + s) * ld];
    }
    return t;
}

// CDNA5 async copy: 16B global -> LDS, tracked by ASYNCcnt (no VGPR round trip).
// GVS addressing: saddr = 64-bit SGPR base, vaddr = 32-bit byte offset,
// "vdst" = VGPR holding LDS byte address.
__device__ __forceinline__ void async_ld16(unsigned lds_byte_addr,
                                           const void* sbase,
                                           unsigned byte_off) {
    asm volatile("global_load_async_to_lds_b128 %0, %1, %2 offset:0"
                 :
                 : "v"(lds_byte_addr), "v"(byte_off), "s"(sbase)
                 : "memory");
}
__device__ __forceinline__ void wait_async0() {
#if __has_builtin(__builtin_amdgcn_s_wait_asynccnt)
    __builtin_amdgcn_s_wait_asynccnt(0);
#else
    asm volatile("s_wait_asynccnt 0x0" ::: "memory");
#endif
}

// ---------------------------------------------------------------------------
// Kernel 0: weight conversion / repack to f16
//   WmgT [64][512]  : WmgT[k][c] = (c<256 ? W_m[c][k] : W_g[c-256][k])
//   WoT  [256][64]  : WoT[k][c]  = W_o[c][k]
// ---------------------------------------------------------------------------
__global__ __launch_bounds__(256) void convert_weights_kernel(
    const float* __restrict__ Wm, const float* __restrict__ Wg,
    const float* __restrict__ Wo,
    _Float16* __restrict__ WmgT, _Float16* __restrict__ WoT) {
    int t = blockIdx.x * blockDim.x + threadIdx.x;
    int stride = gridDim.x * blockDim.x;
    for (int i = t; i < 64 * 512; i += stride) {
        int k = i >> 9, c = i & 511;
        float v = (c < 256) ? Wm[c * 64 + k] : Wg[(c - 256) * 64 + k];
        WmgT[i] = (_Float16)v;
    }
    for (int i = t; i < 256 * 64; i += stride) {
        int k = i >> 6, c = i & 63;
        WoT[i] = (_Float16)Wo[c * 256 + k];
    }
}

// ---------------------------------------------------------------------------
// Kernel 1: LayerNorm over CM=64 for m, wave32 per row (2 elems/lane), f16 out
// ---------------------------------------------------------------------------
__global__ __launch_bounds__(256) void ln_m_kernel(
    const float* __restrict__ m, const float* __restrict__ w,
    const float* __restrict__ b, _Float16* __restrict__ out) {
    int lane = threadIdx.x & 31, wid = threadIdx.x >> 5;
    size_t row = (size_t)blockIdx.x * 8 + wid;
    const float* x = m + row * CM_DIM;
    float x0 = x[lane], x1 = x[lane + 32];
    float s = x0 + x1;
    #pragma unroll
    for (int o = 16; o; o >>= 1) s += __shfl_xor(s, o, 32);
    float mu = s * (1.0f / 64.0f);
    float d0 = x0 - mu, d1 = x1 - mu;
    float q = d0 * d0 + d1 * d1;
    #pragma unroll
    for (int o = 16; o; o >>= 1) q += __shfl_xor(q, o, 32);
    float rs = rsqrtf(q * (1.0f / 64.0f) + 1e-5f);
    out[row * CM_DIM + lane]      = (_Float16)(d0 * rs * w[lane] + b[lane]);
    out[row * CM_DIM + lane + 32] = (_Float16)(d1 * rs * w[lane + 32] + b[lane + 32]);
}

// ---------------------------------------------------------------------------
// Kernel 3: LayerNorm over CZ=128 for z + 8-head bias dot + mask, f32 out
//   bout layout [H][N][N]
// ---------------------------------------------------------------------------
__global__ __launch_bounds__(256) void ln_z_bias_kernel(
    const float* __restrict__ z, const float* __restrict__ nw,
    const float* __restrict__ nb, const float* __restrict__ Wz,
    const unsigned char* __restrict__ mask, float* __restrict__ bout) {
    __shared__ float wz[H_DIM * CZ_DIM];
    for (int i = threadIdx.x; i < H_DIM * CZ_DIM; i += 256) wz[i] = Wz[i];
    __syncthreads();

    int lane = threadIdx.x & 31, wid = threadIdx.x >> 5;
    size_t row = (size_t)blockIdx.x * 8 + wid;          // i*768 + j
    const float* x = z + row * CZ_DIM;

    float v[4];
    float s = 0.f;
    #pragma unroll
    for (int q = 0; q < 4; ++q) { v[q] = x[lane + 32 * q]; s += v[q]; }
    #pragma unroll
    for (int o = 16; o; o >>= 1) s += __shfl_xor(s, o, 32);
    float mu = s * (1.0f / 128.0f);
    float var = 0.f;
    #pragma unroll
    for (int q = 0; q < 4; ++q) { v[q] -= mu; var += v[q] * v[q]; }
    #pragma unroll
    for (int o = 16; o; o >>= 1) var += __shfl_xor(var, o, 32);
    float rs = rsqrtf(var * (1.0f / 128.0f) + 1e-5f);

    float y[4];
    #pragma unroll
    for (int q = 0; q < 4; ++q)
        y[q] = v[q] * rs * nw[lane + 32 * q] + nb[lane + 32 * q];

    float mine = 0.f;
    #pragma unroll
    for (int h = 0; h < H_DIM; ++h) {
        float p = 0.f;
        #pragma unroll
        for (int q = 0; q < 4; ++q) p += y[q] * wz[h * CZ_DIM + lane + 32 * q];
        #pragma unroll
        for (int o = 16; o; o >>= 1) p += __shfl_xor(p, o, 32);
        if (lane == h) mine = p;
    }
    if (lane < H_DIM) {
        float madd = mask[row] ? 0.0f : -1.0e6f;
        bout[(size_t)lane * ROWS_Z + row] = mine + madd;
    }
}

// ---------------------------------------------------------------------------
// Kernel 4: row softmax over j (768), block per (h,i) row, f16 out
// ---------------------------------------------------------------------------
__global__ __launch_bounds__(256) void softmax_kernel(
    const float* __restrict__ bin, _Float16* __restrict__ wout) {
    __shared__ float red[8];
    size_t row = blockIdx.x;
    const float* x = bin + row * N_RES;
    int tid = threadIdx.x, lane = tid & 31, wid = tid >> 5;

    float v[3], mx = -3.4e38f;
    #pragma unroll
    for (int q = 0; q < 3; ++q) { v[q] = x[tid + q * 256]; mx = fmaxf(mx, v[q]); }
    #pragma unroll
    for (int o = 16; o; o >>= 1) mx = fmaxf(mx, __shfl_xor(mx, o, 32));
    if (lane == 0) red[wid] = mx;
    __syncthreads();
    float bm = red[0];
    #pragma unroll
    for (int k = 1; k < 8; ++k) bm = fmaxf(bm, red[k]);
    __syncthreads();

    float e[3], s = 0.f;
    #pragma unroll
    for (int q = 0; q < 3; ++q) { e[q] = expf(v[q] - bm); s += e[q]; }
    #pragma unroll
    for (int o = 16; o; o >>= 1) s += __shfl_xor(s, o, 32);
    if (lane == 0) red[wid] = s;
    __syncthreads();
    float tot = 0.f;
    #pragma unroll
    for (int k = 0; k < 8; ++k) tot += red[k];
    float inv = 1.0f / tot;
    #pragma unroll
    for (int q = 0; q < 3; ++q)
        wout[row * N_RES + tid + q * 256] = (_Float16)(e[q] * inv);
}

// ---------------------------------------------------------------------------
// Tiled WMMA GEMM: C(f32 acc) = A(f16, MxK row-major) * B(f16, KxN row-major)
//   block = 256 threads = 8 waves, wave tile 32x64 (2x4 of 16x16x32 WMMA)
//   BK=32, double-buffered LDS tiles filled with async global->LDS copies
//   (ASYNCcnt), +8-half padding against bank conflicts.
//   EPI 0: split-store v (scatter to [H][j][s*32+d]) + sigmoid g
//   EPI 1: per-head, multiply by g, store tmp f16
//   EPI 2: plain f32 store
// ---------------------------------------------------------------------------
template <int BM, int BN, int WROWS, int WCOLS, int EPI>
__global__ __launch_bounds__(256) void wmma_gemm_kernel(
    const _Float16* __restrict__ A, int lda, long aBatch,
    const _Float16* __restrict__ B, int ldb, long bBatch,
    int K,
    float* __restrict__ outF,
    _Float16* __restrict__ outH,
    _Float16* __restrict__ outH2,
    const _Float16* __restrict__ gIn) {
    constexpr int BK = 32, PAD = 8;
    constexpr int ASZ = BM * (BK + PAD);     // halves per A buffer
    constexpr int BSZ = BK * (BN + PAD);     // halves per B buffer
    __shared__ __align__(16) _Float16 Als[2 * ASZ];
    __shared__ __align__(16) _Float16 Bls[2 * BSZ];

    const int tid  = threadIdx.x;
    const int lane = tid & 31;
    const int wid  = tid >> 5;
    const int wcol = wid % WCOLS;
    const int wrow = wid / WCOLS;
    const int m0 = wrow * 32;
    const int n0 = wcol * 64;

    const int rowBase = blockIdx.y * BM;
    const int colBase = blockIdx.x * BN;
    const int head    = blockIdx.z;
    A += (size_t)head * aBatch;
    B += (size_t)head * bBatch;

    // LDS byte addresses (generic-pointer low 32 bits == LDS offset on CDNA5)
    const unsigned ldsA0 = (unsigned)(size_t)&Als[0];
    const unsigned ldsB0 = (unsigned)(size_t)&Bls[0];

    // async stage of one BK-slice into buffer `buf`
    auto stage = [&](int k0, int buf) {
        constexpr int ACH = BM * BK / 8;     // 8-half (16B) chunks
        #pragma unroll
        for (int it = 0; it < ACH / 256; ++it) {
            int c = it * 256 + tid;
            int r = c / (BK / 8), cc = c % (BK / 8);
            unsigned lds = ldsA0 +
                (unsigned)((buf * ASZ + r * (BK + PAD) + cc * 8) * 2);
            unsigned gof =
                (unsigned)(((unsigned)(rowBase + r) * (unsigned)lda +
                            (unsigned)(k0 + cc * 8)) * 2u);
            async_ld16(lds, A, gof);
        }
        constexpr int BCH = BK * BN / 8;
        #pragma unroll
        for (int it = 0; it < BCH / 256; ++it) {
            int c = it * 256 + tid;
            int r = c / (BN / 8), cc = c % (BN / 8);
            unsigned lds = ldsB0 +
                (unsigned)((buf * BSZ + r * (BN + PAD) + cc * 8) * 2);
            unsigned gof =
                (unsigned)(((unsigned)(k0 + r) * (unsigned)ldb +
                            (unsigned)(colBase + cc * 8)) * 2u);
            async_ld16(lds, B, gof);
        }
    };

    v8f acc[2][4] = {};

    const int laneRow = lane & 15;
    const int kg = lane >> 4;               // A k-group select

    stage(0, 0);
    wait_async0();
    __syncthreads();

    int cur = 0;
    for (int k0 = 0; k0 < K; k0 += BK) {
        if (k0 + BK < K) stage(k0 + BK, cur ^ 1);   // uniform branch

        const _Float16* Ab = &Als[cur * ASZ];
        const _Float16* Bb = &Bls[cur * BSZ];
        v16h bf[4];
        #pragma unroll
        for (int ni = 0; ni < 4; ++ni)
            bf[ni] = load_frag_b(&Bb[n0 + ni * 16 + laneRow], BN + PAD, kg);
        #pragma unroll
        for (int mi = 0; mi < 2; ++mi) {
            v16h af = load_frag_a(
                &Ab[(m0 + mi * 16 + laneRow) * (BK + PAD) + kg * 8]);
            #pragma unroll
            for (int ni = 0; ni < 4; ++ni)
                acc[mi][ni] = __builtin_amdgcn_wmma_f32_16x16x32_f16(
                    false, af, false, bf[ni], (short)0, acc[mi][ni],
                    false, false);
        }
        wait_async0();
        __syncthreads();
        cur ^= 1;
    }

    // ---- epilogue: C layout VGPR e -> M = e + 8*(lane>=16), N = lane&15 ----
    const int mAdd = (lane >> 4) * 8;
    const int nIdx = lane & 15;
    #pragma unroll
    for (int mi = 0; mi < 2; ++mi) {
        #pragma unroll
        for (int ni = 0; ni < 4; ++ni) {
            #pragma unroll
            for (int e = 0; e < 8; ++e) {
                int row = rowBase + m0 + mi * 16 + mAdd + e;
                int col = colBase + n0 + ni * 16 + nIdx;
                float val = acc[mi][ni][e];
                if (EPI == 0) {
                    // row = s*768 + n ; col in [0,512)
                    int s = row / N_RES, n = row % N_RES;
                    if (col < HC_DIM) {
                        int h = col >> 5, d = col & 31;
                        outH[((size_t)(h * N_RES + n)) * VN_DIM + s * CH_DIM + d] =
                            (_Float16)val;
                    } else {
                        outH2[(size_t)row * HC_DIM + (col - HC_DIM)] =
                            (_Float16)(1.0f / (1.0f + expf(-val)));
                    }
                } else if (EPI == 1) {
                    // row = i ; col = s*32 + d ; tmp[(s*768+i)*256 + h*32 + d]
                    int s = col >> 5, d = col & 31;
                    size_t idx = ((size_t)(s * N_RES + row)) * HC_DIM +
                                 head * CH_DIM + d;
                    outH[idx] = (_Float16)(val * (float)gIn[idx]);
                } else {
                    outF[(size_t)row * CM_DIM + col] = val;
                }
            }
        }
    }
}

// ---------------------------------------------------------------------------
// Host launcher
// ---------------------------------------------------------------------------
extern "C" void kernel_launch(void* const* d_in, const int* in_sizes, int n_in,
                              void* d_out, int out_size, void* d_ws, size_t ws_size,
                              hipStream_t stream) {
    (void)in_sizes; (void)n_in; (void)out_size; (void)ws_size;
    const float* m      = (const float*)d_in[0];
    const float* z      = (const float*)d_in[1];
    const unsigned char* mask = (const unsigned char*)d_in[2];
    const float* nmw    = (const float*)d_in[3];
    const float* nmb    = (const float*)d_in[4];
    const float* nzw    = (const float*)d_in[5];
    const float* nzb    = (const float*)d_in[6];
    const float* Wm     = (const float*)d_in[7];
    const float* Wg     = (const float*)d_in[8];
    const float* Wz     = (const float*)d_in[9];
    const float* Wo     = (const float*)d_in[10];
    float* out = (float*)d_out;

    // workspace carve-up (256B aligned)
    char* ws = (char*)d_ws;
    size_t off = 0;
    auto take = [&](size_t bytes) {
        char* p = ws + off;
        off += (bytes + 255) & ~(size_t)255;
        return p;
    };
    _Float16* mln  = (_Float16*)take((size_t)ROWS_M * CM_DIM * 2);        // 25 MB
    _Float16* WmgT = (_Float16*)take((size_t)64 * 512 * 2);
    _Float16* WoT  = (_Float16*)take((size_t)256 * 64 * 2);
    _Float16* vbuf = (_Float16*)take((size_t)H_DIM * N_RES * VN_DIM * 2); // 100 MB
    _Float16* gbuf = (_Float16*)take((size_t)ROWS_M * HC_DIM * 2);        // 100 MB
    float*    bbuf = (float*)   take((size_t)H_DIM * ROWS_Z * 4);         // 19 MB
    _Float16* wbuf = (_Float16*)take((size_t)H_DIM * ROWS_Z * 2);         // 9.4 MB
    _Float16* tmp  = (_Float16*)take((size_t)ROWS_M * HC_DIM * 2);        // 100 MB

    // 0) weights -> f16 packed B matrices
    convert_weights_kernel<<<64, 256, 0, stream>>>(Wm, Wg, Wo, WmgT, WoT);

    // 1) LN(m) -> f16
    ln_m_kernel<<<ROWS_M / 8, 256, 0, stream>>>(m, nmw, nmb, mln);

    // 2) [v|g] = m_ln @ [W_m|W_g]^T   (196608 x 512 x 64), scatter epilogue
    wmma_gemm_kernel<128, 128, 4, 2, 0>
        <<<dim3(512 / 128, ROWS_M / 128, 1), 256, 0, stream>>>(
            mln, CM_DIM, 0, WmgT, 512, 0, CM_DIM,
            nullptr, vbuf, gbuf, nullptr);

    // 3) LN(z) + pair bias + mask -> b[h][i][j] f32
    ln_z_bias_kernel<<<ROWS_Z / 8, 256, 0, stream>>>(z, nzw, nzb, Wz, mask, bbuf);

    // 4) softmax over j -> w f16
    softmax_kernel<<<H_DIM * N_RES, 256, 0, stream>>>(bbuf, wbuf);

    // 5) o[h] = w[h] @ v[h]  (8 x [768 x 8192 x 768]); epilogue *= g -> tmp f16
    wmma_gemm_kernel<128, 128, 4, 2, 1>
        <<<dim3(VN_DIM / 128, N_RES / 128, H_DIM), 256, 0, stream>>>(
            wbuf, N_RES, (long)N_RES * N_RES,
            vbuf, VN_DIM, (long)N_RES * VN_DIM, N_RES,
            nullptr, tmp, nullptr, gbuf);

    // 6) out = tmp @ W_o^T  (196608 x 64 x 256) -> f32
    wmma_gemm_kernel<256, 64, 8, 1, 2>
        <<<dim3(1, ROWS_M / 256, 1), 256, 0, stream>>>(
            tmp, HC_DIM, 0, WoT, CM_DIM, 0, HC_DIM,
            out, nullptr, nullptr, nullptr);
}
